// DeformableTransformer_84095459656068
// MI455X (gfx1250) — matrix operand order repaired and checked by reading, not compiled, not run on hardware
//
#include <hip/hip_runtime.h>
#include <hip/hip_bf16.h>

typedef _Float16 v16h __attribute__((ext_vector_type(16)));
typedef _Float16 h8   __attribute__((ext_vector_type(8)));
typedef float    v8f  __attribute__((ext_vector_type(8)));

// ---------------- constants ----------------
#define BB   8
#define LQ   300
#define DM   256
#define DFFN 2048
#define NH   8
#define DH   32
#define SFULL 3840          // 2048+1024+512+256
#define TQ   (BB*LQ)        // 2400 query tokens
#define BH   (BB*NH)        // 64
#define LKP  320            // padded key length (mult of 32)
#define LRP  304            // padded query rows (mult of 16)
#define HLP  128

#define LDS_STRIDE 40       // halves per LDS tile row (32 data + 8 pad -> 80B, bank-spread)

// LDS byte-offset of a __shared__ object (addrspace(3) pointers are 32-bit offsets)
__device__ __forceinline__ unsigned lds_off(const void* p)
{
    return (unsigned)(unsigned long long)(const __attribute__((address_space(3))) char*)p;
}

__device__ __forceinline__ void async_b128(unsigned ldsoff, const void* gaddr)
{
    // GLOBAL_LOAD_ASYNC_TO_LDS_B128: VDST = LDS addr VGPR, VADDR = 64-bit addr, no saddr
    asm volatile("global_load_async_to_lds_b128 %0, %1, off"
                 :: "v"(ldsoff), "v"(gaddr) : "memory");
}

// =====================================================================
// Staged WMMA GEMM for contiguous-B ("weight style") operands.
//   C[M,N] = A[M,K] * W^T + bias,  W element (n,k) at Bw[n*ldbn + k]
//   Block: 256 thr / 8 waves, macro tile 128x64, wave tile 32x32.
//   A/B K-slabs double-buffered in LDS via global_load_async_to_lds_b128,
//   synchronized with s_wait_asynccnt + barrier (ASYNCcnt protocol).
//   Requires: K % 32 == 0. M,N guarded by clamp (load) + mask (store).
// =====================================================================
__global__ __launch_bounds__(256) void gemm_wmma_staged_kernel(
    const _Float16* __restrict__ A, long sAi, long sAo, int nbIn, int lda,
    const _Float16* __restrict__ Bw, long sBi, long sBo, int ldbn,
    const float* __restrict__ bias,
    float* __restrict__ Cf, _Float16* __restrict__ Ch, long sCi, long sCo, int ldc,
    int M, int N, int K, int tilesN, int relu)
{
    __shared__ _Float16 sA[2][128 * LDS_STRIDE];
    __shared__ _Float16 sB[2][64 * LDS_STRIDE];

    const int tid  = threadIdx.x;
    const int wave = tid >> 5;
    const int lane = tid & 31;
    const int l16  = lane & 15;
    const int hi   = lane >> 4;

    const int bt = blockIdx.y;
    const int bo = bt / nbIn, bi = bt % nbIn;
    const int tm = blockIdx.x / tilesN;
    const int tn = blockIdx.x % tilesN;
    const int m0 = tm * 128, n0 = tn * 64;

    const _Float16* Ab = A  + (long)bo * sAo + (long)bi * sAi;
    const _Float16* Bb = Bw + (long)bo * sBo + (long)bi * sBi;

    // ---- staging map: thread -> 16B chunks ----
    const int crow = tid >> 2;                 // 0..63
    const int cc   = (tid & 3) * 8;            // half offset within 32-K slab
    const int ga0  = (m0 + crow      < M) ? (m0 + crow)      : (M - 1);
    const int ga1  = (m0 + crow + 64 < M) ? (m0 + crow + 64) : (M - 1);
    const int gb0  = (n0 + crow      < N) ? (n0 + crow)      : (N - 1);

    const int mw = wave & 3;                   // 0..3  -> wave row block
    const int nw = wave >> 2;                  // 0..1  -> wave col block

    v8f c00 = {0.f,0.f,0.f,0.f,0.f,0.f,0.f,0.f};
    v8f c01 = c00, c10 = c00, c11 = c00;

    auto stage = [&](int buf, int kt) {
        async_b128(lds_off(&sA[buf][crow * LDS_STRIDE + cc]),
                   Ab + (long)ga0 * lda + kt + cc);
        async_b128(lds_off(&sA[buf][(crow + 64) * LDS_STRIDE + cc]),
                   Ab + (long)ga1 * lda + kt + cc);
        async_b128(lds_off(&sB[buf][crow * LDS_STRIDE + cc]),
                   Bb + (long)gb0 * ldbn + kt + cc);
    };

    int buf = 0;
    stage(0, 0);
    for (int kt = 0; kt < K; kt += 32) {
        const bool more = (kt + 32) < K;
        if (more) stage(buf ^ 1, kt + 32);
        if (more) asm volatile("s_wait_asynccnt 0x3" ::: "memory");
        else      asm volatile("s_wait_asynccnt 0x0" ::: "memory");
        __syncthreads();

        // ---- fragments from LDS (contiguous k = 16*hi + s, same for A and B) ----
        v16h a0, a1, b0, b1;
        {
            const int r0 = mw * 32 + l16;
            const h8 x0 = *(const h8*)&sA[buf][ r0       * LDS_STRIDE + hi * 16];
            const h8 x1 = *(const h8*)&sA[buf][ r0       * LDS_STRIDE + hi * 16 + 8];
            const h8 x2 = *(const h8*)&sA[buf][(r0 + 16) * LDS_STRIDE + hi * 16];
            const h8 x3 = *(const h8*)&sA[buf][(r0 + 16) * LDS_STRIDE + hi * 16 + 8];
            const int n = nw * 32 + l16;
            const h8 y0 = *(const h8*)&sB[buf][ n        * LDS_STRIDE + hi * 16];
            const h8 y1 = *(const h8*)&sB[buf][ n        * LDS_STRIDE + hi * 16 + 8];
            const h8 y2 = *(const h8*)&sB[buf][(n + 16)  * LDS_STRIDE + hi * 16];
            const h8 y3 = *(const h8*)&sB[buf][(n + 16)  * LDS_STRIDE + hi * 16 + 8];
#pragma unroll
            for (int j = 0; j < 8; ++j) {
                a0[j] = x0[j]; a0[8 + j] = x1[j];
                a1[j] = x2[j]; a1[8 + j] = x3[j];
                b0[j] = y0[j]; b0[8 + j] = y1[j];
                b1[j] = y2[j]; b1[8 + j] = y3[j];
            }
        }
        c00 = __builtin_amdgcn_wmma_f32_16x16x32_f16(false, a0, false, b0, (short)0, c00, false, false);
        c01 = __builtin_amdgcn_wmma_f32_16x16x32_f16(false, a0, false, b1, (short)0, c01, false, false);
        c10 = __builtin_amdgcn_wmma_f32_16x16x32_f16(false, a1, false, b0, (short)0, c10, false, false);
        c11 = __builtin_amdgcn_wmma_f32_16x16x32_f16(false, a1, false, b1, (short)0, c11, false, false);
        __syncthreads();
        buf ^= 1;
    }

    // ---- store 4 tiles: elem r -> (row + r + 8*hi, colbase + l16) ----
    float*    cfp = Cf ? (Cf + (long)bo * sCo + (long)bi * sCi) : (float*)0;
    _Float16* chp = Ch ? (Ch + (long)bo * sCo + (long)bi * sCi) : (_Float16*)0;
    const int rb = m0 + mw * 32;
    const int cb = n0 + nw * 32;
#pragma unroll
    for (int ti = 0; ti < 4; ++ti) {
        const v8f  c   = (ti == 0) ? c00 : (ti == 1) ? c01 : (ti == 2) ? c10 : c11;
        const int  rowb = rb + ((ti >> 1) ? 16 : 0);
        const int  col  = cb + ((ti & 1) ? 16 : 0) + l16;
        if (col < N) {
            const float badd = bias ? bias[col] : 0.f;
#pragma unroll
            for (int r = 0; r < 8; ++r) {
                const int row = rowb + r + hi * 8;
                if (row < M) {
                    float v = c[r] + badd;
                    if (relu) v = v > 0.f ? v : 0.f;
                    const long idx = (long)row * ldc + col;
                    if (cfp) cfp[idx] = v;
                    if (chp) chp[idx] = (_Float16)v;
                }
            }
        }
    }
}

// =====================================================================
// Generic direct-global WMMA GEMM (kept only for strided-B P@V).
// =====================================================================
__global__ __launch_bounds__(256) void gemm_wmma_kernel(
    const _Float16* __restrict__ A, long sAi, long sAo, int nbIn, int lda,
    const _Float16* __restrict__ Bw, long sBi, long sBo, int ldbn, int ldbk,
    const float* __restrict__ bias,
    float* __restrict__ Cf, _Float16* __restrict__ Ch, long sCi, long sCo, int ldc,
    int M, int N, int Kloop, int Kvalid,
    int batches, int tilesM, int tilesN, int relu)
{
    const int wave = threadIdx.x >> 5;
    const int lane = threadIdx.x & 31;
    const long perB = (long)tilesM * tilesN;
    const long tile = (long)blockIdx.x * 8 + wave;
    if (tile >= (long)batches * perB) return;    // uniform per wave

    const int bt  = (int)(tile / perB);
    const int rem = (int)(tile % perB);
    const int tm  = rem / tilesN;
    const int tn  = rem % tilesN;
    const int bo  = bt / nbIn;
    const int bi  = bt % nbIn;

    const _Float16* Ab = A  + (long)bo * sAo + (long)bi * sAi;
    const _Float16* Bb = Bw + (long)bo * sBo + (long)bi * sBi;

    const int l16  = lane & 15;
    const int hi   = lane >> 4;
    const int arow = tm * 16 + l16;
    const int bn   = tn * 16 + l16;
    const bool aval = (arow < M);
    const bool bval = (bn < N);

    v8f c = {0.f,0.f,0.f,0.f,0.f,0.f,0.f,0.f};

    for (int k = 0; k < Kloop; k += 32) {
        v16h a, b;
#pragma unroll
        for (int j = 0; j < 16; ++j) { a[j] = (_Float16)0.f; b[j] = (_Float16)0.f; }

        if (aval) {
            const _Float16* ap = Ab + (long)arow * lda + k;
            const h8 x0 = *(const h8*)(ap + hi * 16);
            const h8 x1 = *(const h8*)(ap + hi * 16 + 8);
#pragma unroll
            for (int j = 0; j < 8; ++j) { a[j] = x0[j]; a[8 + j] = x1[j]; }
        }
        const int kb = k + hi * 16;
        if (bval) {
            if (ldbk == 1 && (kb + 15) < Kvalid) {
                const _Float16* bp = Bb + (long)bn * ldbn + kb;
                const h8 y0 = *(const h8*)bp;
                const h8 y1 = *(const h8*)(bp + 8);
#pragma unroll
                for (int j = 0; j < 8; ++j) { b[j] = y0[j]; b[8 + j] = y1[j]; }
                __builtin_prefetch(bp + 32, 0, 1);
            } else {
#pragma unroll
                for (int j = 0; j < 16; ++j) {
                    const int kk = kb + j;
                    b[j] = (kk < Kvalid) ? Bb[(long)bn * ldbn + (long)kk * ldbk]
                                         : (_Float16)0.f;
                }
            }
        }
        c = __builtin_amdgcn_wmma_f32_16x16x32_f16(
                false, a, false, b, (short)0, c, false, false);
    }

    const int col = tn * 16 + l16;
    if (col < N) {
        const float badd = bias ? bias[col] : 0.f;
        float*    cfp = Cf ? (Cf + (long)bo * sCo + (long)bi * sCi) : (float*)0;
        _Float16* chp = Ch ? (Ch + (long)bo * sCo + (long)bi * sCi) : (_Float16*)0;
#pragma unroll
        for (int r = 0; r < 8; ++r) {
            const int row = tm * 16 + r + hi * 8;
            if (row < M) {
                float v = c[r] + badd;
                if (relu) v = v > 0.f ? v : 0.f;
                const long idx = (long)row * ldc + col;
                if (cfp) cfp[idx] = v;
                if (chp) chp[idx] = (_Float16)v;
            }
        }
    }
}

// ---------------- elementwise helpers ----------------
__global__ void f32_to_f16_kernel(const float* __restrict__ in,
                                  _Float16* __restrict__ out, long n)
{
    long i = (long)blockIdx.x * blockDim.x + threadIdx.x;
    if (i < n) out[i] = (_Float16)in[i];
}

__global__ void addpos_kernel(const float* __restrict__ X, const float* __restrict__ pos,
                              _Float16* __restrict__ Qh, _Float16* __restrict__ Xh, long n)
{
    long i = (long)blockIdx.x * blockDim.x + threadIdx.x;
    if (i < n) {
        float x = X[i];
        Qh[i] = (_Float16)(x + pos[i]);
        Xh[i] = (_Float16)x;
    }
}

// ---------------- residual + layernorm (block = one token, 256 threads) ----
__global__ __launch_bounds__(256) void ln_kernel(
    float* __restrict__ X, const float* __restrict__ T,
    const float* __restrict__ g, const float* __restrict__ bta,
    _Float16* __restrict__ Xh)
{
    __shared__ float red[256];
    const int t = blockIdx.x, i = threadIdx.x;
    const long idx = (long)t * DM + i;
    const float x = X[idx] + T[idx];
    red[i] = x; __syncthreads();
    for (int s = 128; s > 0; s >>= 1) { if (i < s) red[i] += red[i + s]; __syncthreads(); }
    const float mean = red[0] * (1.f / 256.f); __syncthreads();
    const float d = x - mean;
    red[i] = d * d; __syncthreads();
    for (int s = 128; s > 0; s >>= 1) { if (i < s) red[i] += red[i + s]; __syncthreads(); }
    const float var = red[0] * (1.f / 256.f);
    const float y = d * rsqrtf(var + 1e-5f) * g[i] + bta[i];
    X[idx]  = y;
    Xh[idx] = (_Float16)y;
}

// ---------------- self-attention softmax (row of 300, padded to 320) -------
__global__ __launch_bounds__(128) void sa_softmax_kernel(
    const float* __restrict__ scores, _Float16* __restrict__ P)
{
    __shared__ float red[128];
    const int row = blockIdx.x;          // 0..299
    const int bh  = blockIdx.y;          // 0..63
    const int tid = threadIdx.x;
    const float scale = 0.17677669529663687f;   // 1/sqrt(32)
    const long base = (long)bh * (LRP * LKP) + (long)row * LKP;

    float m = -3.4e38f;
    for (int c = tid; c < LQ; c += 128) m = fmaxf(m, scores[base + c] * scale);
    red[tid] = m; __syncthreads();
    for (int s = 64; s > 0; s >>= 1) { if (tid < s) red[tid] = fmaxf(red[tid], red[tid + s]); __syncthreads(); }
    m = red[0]; __syncthreads();

    float sum = 0.f;
    for (int c = tid; c < LQ; c += 128) sum += __expf(scores[base + c] * scale - m);
    red[tid] = sum; __syncthreads();
    for (int s = 64; s > 0; s >>= 1) { if (tid < s) red[tid] += red[tid + s]; __syncthreads(); }
    const float inv = 1.f / red[0];

    for (int c = tid; c < LKP; c += 128)
        P[base + c] = (c < LQ) ? (_Float16)(__expf(scores[base + c] * scale - m) * inv)
                               : (_Float16)0.f;
}

// ---------------- deformable-attn softmax over 16 (Lv*P) per head ----------
__global__ void ca_softmax_kernel(const float* __restrict__ logits,
                                  float* __restrict__ attnW)
{
    const int g = blockIdx.x * blockDim.x + threadIdx.x;
    if (g >= TQ * NH) return;
    const long base = (long)(g >> 3) * HLP + (long)(g & 7) * 16;
    float m = -3.4e38f;
#pragma unroll
    for (int j = 0; j < 16; ++j) m = fmaxf(m, logits[base + j]);
    float e[16], s = 0.f;
#pragma unroll
    for (int j = 0; j < 16; ++j) { e[j] = __expf(logits[base + j] - m); s += e[j]; }
    const float inv = 1.f / s;
#pragma unroll
    for (int j = 0; j < 16; ++j) attnW[base + j] = e[j] * inv;
}

// ---------------- deformable temporal bilinear sampling --------------------
__global__ __launch_bounds__(256) void ca_sample_kernel(
    const _Float16* __restrict__ valH, const float* __restrict__ offF,
    const float* __restrict__ attnW, const float* __restrict__ refpts,
    const float* __restrict__ ratios, _Float16* __restrict__ sampH)
{
    __shared__ float s_off[HLP], s_att[HLP];
    const int t   = blockIdx.x;          // 0..TQ-1
    const int tid = threadIdx.x;
    const int h   = tid >> 5, dh = tid & 31;
    const int b   = t / LQ;
    if (tid < HLP) {
        s_off[tid] = offF [(long)t * HLP + tid];
        s_att[tid] = attnW[(long)t * HLP + tid];
    }
    __syncthreads();

    const int   Ts[4] = {2048, 1024, 512, 256};
    const int   st[4] = {0, 2048, 3072, 3584};
    const float ref = refpts[t];
    const _Float16* vb = valH + (long)b * SFULL * DM + h * DH + dh;

    float acc = 0.f;
#pragma unroll
    for (int lvl = 0; lvl < 4; ++lvl) {
        const float T  = (float)Ts[lvl];
        const float rp = ref * ratios[b * 4 + lvl];
#pragma unroll
        for (int p = 0; p < 4; ++p) {
            const int idx = h * 16 + lvl * 4 + p;
            const float off = s_off[idx];
            const float att = s_att[idx];
            const float pos = (rp + off / T) * T - 0.5f;
            const float fl  = floorf(pos);
            const float w   = pos - fl;
            const int i0 = (int)fl;
            const int i1 = i0 + 1;
            const float v0 = (i0 >= 0 && i0 < Ts[lvl]) ? (float)vb[(long)(st[lvl] + i0) * DM] : 0.f;
            const float v1 = (i1 >= 0 && i1 < Ts[lvl]) ? (float)vb[(long)(st[lvl] + i1) * DM] : 0.f;
            acc += att * (v0 * (1.f - w) + v1 * w);
        }
    }
    sampH[(long)t * DM + tid] = (_Float16)acc;
}

// =====================================================================
// host side
// =====================================================================
static inline void gemm_staged(hipStream_t st,
    const _Float16* A, long sAi, long sAo, int nbIn, int lda,
    const _Float16* Bw, long sBi, long sBo, int ldbn,
    const float* bias, float* Cf, _Float16* Ch, long sCi, long sCo, int ldc,
    int M, int N, int K, int batches, int relu)
{
    const int tm = (M + 127) / 128, tn = (N + 63) / 64;
    dim3 grid(tm * tn, batches);
    gemm_wmma_staged_kernel<<<grid, 256, 0, st>>>(A, sAi, sAo, nbIn, lda,
        Bw, sBi, sBo, ldbn, bias, Cf, Ch, sCi, sCo, ldc, M, N, K, tn, relu);
}

static inline void gemm(hipStream_t st,
    const _Float16* A, long sAi, long sAo, int nbIn, int lda,
    const _Float16* Bw, long sBi, long sBo, int ldbn, int ldbk,
    const float* bias, float* Cf, _Float16* Ch, long sCi, long sCo, int ldc,
    int M, int N, int Kloop, int Kvalid, int batches, int relu)
{
    const int tm = (M + 15) / 16, tn = (N + 15) / 16;
    const long total = (long)batches * tm * tn;
    const int blocks = (int)((total + 7) / 8);
    gemm_wmma_kernel<<<blocks, 256, 0, st>>>(A, sAi, sAo, nbIn, lda,
        Bw, sBi, sBo, ldbn, ldbk, bias, Cf, Ch, sCi, sCo, ldc,
        M, N, Kloop, Kvalid, batches, tm, tn, relu);
}

static inline void cvt(hipStream_t st, const float* in, _Float16* out, long n)
{
    f32_to_f16_kernel<<<(int)((n + 255) / 256), 256, 0, st>>>(in, out, n);
}

extern "C" void kernel_launch(void* const* d_in, const int* in_sizes, int n_in,
                              void* d_out, int out_size, void* d_ws, size_t ws_size,
                              hipStream_t stream)
{
    (void)in_sizes; (void)n_in; (void)out_size; (void)ws_size;
    const int L = 6;

    const float* tgt      = (const float*)d_in[0];
    const float* qpos     = (const float*)d_in[1];
    const float* refp     = (const float*)d_in[2];
    const float* src      = (const float*)d_in[3];
    const float* ratios   = (const float*)d_in[4];
    const float* ca_val_w = (const float*)d_in[5];
    const float* ca_val_b = (const float*)d_in[6];
    const float* ca_off_w = (const float*)d_in[7];
    const float* ca_off_b = (const float*)d_in[8];
    const float* ca_att_w = (const float*)d_in[9];
    const float* ca_att_b = (const float*)d_in[10];
    const float* ca_out_w = (const float*)d_in[11];
    const float* ca_out_b = (const float*)d_in[12];
    const float* sa_in_w  = (const float*)d_in[13];
    const float* sa_in_b  = (const float*)d_in[14];
    const float* sa_out_w = (const float*)d_in[15];
    const float* sa_out_b = (const float*)d_in[16];
    const float* ffn_w1   = (const float*)d_in[17];
    const float* ffn_b1   = (const float*)d_in[18];
    const float* ffn_w2   = (const float*)d_in[19];
    const float* ffn_b2   = (const float*)d_in[20];
    const float* ln1_g = (const float*)d_in[21];
    const float* ln1_b = (const float*)d_in[22];
    const float* ln2_g = (const float*)d_in[23];
    const float* ln2_b = (const float*)d_in[24];
    const float* ln3_g = (const float*)d_in[25];
    const float* ln3_b = (const float*)d_in[26];

    // ---- workspace carve ----
    char* base = (char*)d_ws;
    size_t off = 0;
    auto alloc = [&](size_t bytes) -> void* {
        void* r = base + off;
        off += (bytes + 255) & ~(size_t)255;
        return r;
    };
    const long nXD = (long)TQ * DM;
    float*    X      = (float*)   alloc(nXD * 4);
    float*    t2     = (float*)   alloc(nXD * 4);
    float*    scoresB= (float*)   alloc((long)BH * LRP * LKP * 4);
    float*    offF   = (float*)   alloc((long)TQ * HLP * 4);
    float*    attF   = (float*)   alloc((long)TQ * HLP * 4);
    float*    attW   = (float*)   alloc((long)TQ * HLP * 4);
    _Float16* Xh     = (_Float16*)alloc(nXD * 2);
    _Float16* Qh     = (_Float16*)alloc(nXD * 2);
    _Float16* qkb    = (_Float16*)alloc((long)TQ * 512 * 2);
    _Float16* vbuf   = (_Float16*)alloc(nXD * 2);
    _Float16* Pb     = (_Float16*)alloc((long)BH * LRP * LKP * 2);
    _Float16* attnH  = (_Float16*)alloc(nXD * 2);
    _Float16* sampH  = (_Float16*)alloc(nXD * 2);
    _Float16* H1h    = (_Float16*)alloc((long)TQ * DFFN * 2);
    _Float16* srcH   = (_Float16*)alloc((long)BB * SFULL * DM * 2);
    _Float16* valH   = (_Float16*)alloc((long)BB * SFULL * DM * 2);
    _Float16* wSAin  = (_Float16*)alloc((long)L * 768 * DM * 2);
    _Float16* wSAout = (_Float16*)alloc((long)L * DM * DM * 2);
    _Float16* wCAval = (_Float16*)alloc((long)L * DM * DM * 2);
    _Float16* wCAoff = (_Float16*)alloc((long)L * HLP * DM * 2);
    _Float16* wCAatt = (_Float16*)alloc((long)L * HLP * DM * 2);
    _Float16* wCAout = (_Float16*)alloc((long)L * DM * DM * 2);
    _Float16* wF1    = (_Float16*)alloc((long)L * DFFN * DM * 2);
    _Float16* wF2    = (_Float16*)alloc((long)L * DM * DFFN * 2);

    // ---- one-time conversions ----
    cvt(stream, src,      srcH,   (long)BB * SFULL * DM);
    cvt(stream, sa_in_w,  wSAin,  (long)L * 768 * DM);
    cvt(stream, sa_out_w, wSAout, (long)L * DM * DM);
    cvt(stream, ca_val_w, wCAval, (long)L * DM * DM);
    cvt(stream, ca_off_w, wCAoff, (long)L * HLP * DM);
    cvt(stream, ca_att_w, wCAatt, (long)L * HLP * DM);
    cvt(stream, ca_out_w, wCAout, (long)L * DM * DM);
    cvt(stream, ffn_w1,   wF1,    (long)L * DFFN * DM);
    cvt(stream, ffn_w2,   wF2,    (long)L * DM * DFFN);

    hipMemcpyAsync(X, tgt, nXD * sizeof(float), hipMemcpyDeviceToDevice, stream);

    const int nblk = (int)((nXD + 255) / 256);

    for (int l = 0; l < L; ++l) {
        const _Float16* wsa  = wSAin  + (long)l * 768 * DM;
        const _Float16* wsao = wSAout + (long)l * DM * DM;
        const _Float16* wcv  = wCAval + (long)l * DM * DM;
        const _Float16* wco  = wCAoff + (long)l * HLP * DM;
        const _Float16* wca  = wCAatt + (long)l * HLP * DM;
        const _Float16* wcoo = wCAout + (long)l * DM * DM;
        const _Float16* w1   = wF1    + (long)l * DFFN * DM;
        const _Float16* w2   = wF2    + (long)l * DM * DFFN;

        // ================= self-attention =================
        addpos_kernel<<<nblk, 256, 0, stream>>>(X, qpos, Qh, Xh, nXD);
        gemm_staged(stream, Qh, 0, 0, 1, DM, wsa, 0, 0, DM,
                    sa_in_b + (long)l * 768, nullptr, qkb, 0, 0, 512,
                    TQ, 512, DM, 1, 0);
        gemm_staged(stream, Xh, 0, 0, 1, DM, wsa + 512 * DM, 0, 0, DM,
                    sa_in_b + (long)l * 768 + 512, nullptr, vbuf, 0, 0, DM,
                    TQ, DM, DM, 1, 0);
        // scores = q @ k^T per (b,h): B = k rows (contiguous K) -> staged kernel
        gemm_staged(stream, qkb, 32, (long)LQ * 512, NH, 512,
                    qkb + 256, 32, (long)LQ * 512, 512,
                    nullptr, scoresB, nullptr, (long)LRP * LKP, (long)NH * LRP * LKP, LKP,
                    LQ, LQ, DH, BH, 0);
        sa_softmax_kernel<<<dim3(LQ, BH), 128, 0, stream>>>(scoresB, Pb);
        // out = P @ V per (b,h)  (strided B -> generic kernel)
        gemm(stream, Pb, (long)LRP * LKP, (long)NH * LRP * LKP, NH, LKP,
             vbuf, 32, (long)LQ * DM, 1, DM,
             nullptr, nullptr, attnH, 32, (long)LQ * DM, DM,
             LQ, DH, LKP, LQ, BH, 0);
        gemm_staged(stream, attnH, 0, 0, 1, DM, wsao, 0, 0, DM,
                    sa_out_b + (long)l * DM, t2, nullptr, 0, 0, DM,
                    TQ, DM, DM, 1, 0);
        ln_kernel<<<TQ, 256, 0, stream>>>(X, t2, ln2_g + (long)l * DM, ln2_b + (long)l * DM, Xh);

        // ================= deformable cross-attention =================
        addpos_kernel<<<nblk, 256, 0, stream>>>(X, qpos, Qh, Xh, nXD);
        gemm_staged(stream, srcH, 0, 0, 1, DM, wcv, 0, 0, DM,
                    ca_val_b + (long)l * DM, nullptr, valH, 0, 0, DM,
                    BB * SFULL, DM, DM, 1, 0);
        gemm_staged(stream, Qh, 0, 0, 1, DM, wco, 0, 0, DM,
                    ca_off_b + (long)l * HLP, offF, nullptr, 0, 0, HLP,
                    TQ, HLP, DM, 1, 0);
        gemm_staged(stream, Qh, 0, 0, 1, DM, wca, 0, 0, DM,
                    ca_att_b + (long)l * HLP, attF, nullptr, 0, 0, HLP,
                    TQ, HLP, DM, 1, 0);
        ca_softmax_kernel<<<(TQ * NH + 63) / 64, 64, 0, stream>>>(attF, attW);
        ca_sample_kernel<<<TQ, 256, 0, stream>>>(valH, offF, attW, refp, ratios, sampH);
        gemm_staged(stream, sampH, 0, 0, 1, DM, wcoo, 0, 0, DM,
                    ca_out_b + (long)l * DM, t2, nullptr, 0, 0, DM,
                    TQ, DM, DM, 1, 0);
        ln_kernel<<<TQ, 256, 0, stream>>>(X, t2, ln1_g + (long)l * DM, ln1_b + (long)l * DM, Xh);

        // ================= FFN =================
        gemm_staged(stream, Xh, 0, 0, 1, DM, w1, 0, 0, DM,
                    ffn_b1 + (long)l * DFFN, nullptr, H1h, 0, 0, DFFN,
                    TQ, DFFN, DM, 1, 1 /*relu*/);
        gemm_staged(stream, H1h, 0, 0, 1, DFFN, w2, 0, 0, DFFN,
                    ffn_b2 + (long)l * DM, t2, nullptr, 0, 0, DM,
                    TQ, DM, DFFN, 1, 0);
        ln_kernel<<<TQ, 256, 0, stream>>>(X, t2, ln3_g + (long)l * DM, ln3_b + (long)l * DM, Xh);
    }

    hipMemcpyAsync(d_out, X, nXD * sizeof(float), hipMemcpyDeviceToDevice, stream);
}
